// SLiMPerformer_35201551958116
// MI455X (gfx1250) — matrix operand order repaired and checked by reading, not compiled, not run on hardware
//
#include <hip/hip_runtime.h>
#include <math.h>

// Model dims (fixed by the reference)
#define BB   2
#define TT   2048
#define DD   512
#define LL   2
#define FF   2048
#define HH   8
#define HDm  64
#define VV   256
#define NTOK (BB*TT)   // 4096 rows
#define CH   64        // attention chunk length
#define NC   (TT/CH)   // 32 chunks per sequence

typedef __bf16 bf16_t;
typedef __attribute__((ext_vector_type(16))) __bf16        v16bf;
typedef __attribute__((ext_vector_type(8)))  float         v8f;
typedef __attribute__((ext_vector_type(4)))  unsigned int  u32x4;
typedef __attribute__((ext_vector_type(8)))  unsigned int  u32x8;

__device__ __forceinline__ bf16_t f2bf(float f) { return (bf16_t)f; }

__device__ __forceinline__ v16bf pack16(u32x4 lo, u32x4 hi)
{
    u32x8 r;
    r[0] = lo[0]; r[1] = lo[1]; r[2] = lo[2]; r[3] = lo[3];
    r[4] = hi[0]; r[5] = hi[1]; r[6] = hi[2]; r[7] = hi[3];
    return __builtin_bit_cast(v16bf, r);
}

// A-operand (16x32 bf16) from a row-major row pointer: per-lane K contiguous (k = 16*half + s).
__device__ __forceinline__ v16bf a_op(const bf16_t* row, int k0, int half)
{
    u32x4 lo = *(const u32x4*)(row + k0 + (half ? 16 : 0));
    u32x4 hi = *(const u32x4*)(row + k0 + (half ? 24 : 8));
    return pack16(lo, hi);
}

// B-operand (32x16 bf16) where per-lane K values are contiguous at `row`.
__device__ __forceinline__ v16bf b_op(const bf16_t* row, int k0, int half)
{
    const bf16_t* p = row + k0 + (half ? 16 : 0);
    u32x4 lo = *(const u32x4*)p;
    u32x4 hi = *(const u32x4*)(p + 8);
    return pack16(lo, hi);
}

#define WMMA_BF16(av, bv, c) __builtin_amdgcn_wmma_f32_16x16x32_bf16( \
        false, (av), false, (bv), (short)0, (c), false, false)

// Async global->LDS copy of 16 bytes (CDNA5, ASYNCcnt-tracked).
__device__ __forceinline__ void async_cp_b128(unsigned lds_off, const void* gptr)
{
    asm volatile("global_load_async_to_lds_b128 %0, %1, off"
                 :: "v"(lds_off), "v"((unsigned long long)gptr) : "memory");
}
__device__ __forceinline__ void async_wait0()
{
    asm volatile("s_wait_asynccnt 0x0" ::: "memory");
}

// ---------------------------------------------------------------------------
// Pack a row-major K x N fp32 weight matrix into bf16 WMMA B-operand order.
// ---------------------------------------------------------------------------
__global__ void pack_w(const float* __restrict__ W, bf16_t* __restrict__ P,
                       int K, int N, float scale)
{
    int idx = blockIdx.x * blockDim.x + threadIdx.x;
    if (idx >= K * N) return;
    int e    = idx & 15;
    int lane = (idx >> 4) & 31;
    int kb   = (idx >> 9) % (K >> 5);
    int nt   = idx / ((K >> 5) << 9);
    int k = (kb << 5) + ((lane & 16) ? 16 : 0) + e;
    int n = (nt << 4) + (lane & 15);
    P[idx] = f2bf(W[(size_t)k * N + n] * scale);
}

// ---------------------------------------------------------------------------
// token embedding + sinusoidal positional concat -> x (f32) + xb (bf16)
// ---------------------------------------------------------------------------
__global__ void embed_pos(const int* __restrict__ tok, const float* __restrict__ emb,
                          float* __restrict__ x, bf16_t* __restrict__ xb)
{
    int idx = blockIdx.x * blockDim.x + threadIdx.x;
    if (idx >= NTOK * DD) return;
    int bt = idx >> 9;
    int c  = idx & (DD - 1);
    int t  = bt & (TT - 1);
    float val;
    if (c < 256) {
        val = emb[(size_t)tok[bt] * 256 + c];
    } else if (c < 384) {
        int j = c - 256;
        float f = expf(-9.210340371976184f * (float)(2 * j) * (1.0f / 256.0f));
        val = sinf((float)t * f);
    } else {
        int j = c - 384;
        float f = expf(-9.210340371976184f * (float)(2 * j) * (1.0f / 256.0f));
        val = cosf((float)t * f);
    }
    x[idx]  = val;
    xb[idx] = f2bf(val);
}

// ---------------------------------------------------------------------------
// bf16 WMMA GEMM with pre-packed B. Block = 4 waves; wave = 16(M)x64(N) strip.
// ---------------------------------------------------------------------------
__global__ __launch_bounds__(128)
void gemm_bf16_wmma(const bf16_t* __restrict__ A, int lda,
                    const bf16_t* __restrict__ Bp,
                    const float* __restrict__ bias,
                    float* __restrict__ Cf, bf16_t* __restrict__ Cb, int ldc,
                    int K, int act)
{
    const int lane    = threadIdx.x & 31;
    const int wave    = threadIdx.x >> 5;
    const int half    = lane >> 4;
    const int l16     = lane & 15;
    const int tile_m  = blockIdx.y * 64 + wave * 16;
    const int tile_n0 = blockIdx.x * 64;
    const int kbc     = K >> 5;

    v8f acc[4];
    #pragma unroll
    for (int s = 0; s < 4; ++s) acc[s] = (v8f){0.f,0.f,0.f,0.f,0.f,0.f,0.f,0.f};

    const bf16_t* arow  = A + (size_t)(tile_m + l16) * lda;
    const bf16_t* bbase = Bp + ((size_t)(tile_n0 >> 4) * kbc) * 512 + lane * 16;
    const size_t  bsub  = (size_t)kbc * 512;

    for (int kb = 0; kb < kbc; ++kb) {
        v16bf av = a_op(arow, kb << 5, half);
        const bf16_t* bp = bbase + (size_t)kb * 512;
        #pragma unroll
        for (int s = 0; s < 4; ++s) {
            u32x4 lo = *(const u32x4*)(bp + (size_t)s * bsub);
            u32x4 hi = *(const u32x4*)(bp + (size_t)s * bsub + 8);
            acc[s] = WMMA_BF16(av, pack16(lo, hi), acc[s]);
        }
    }

    #pragma unroll
    for (int s = 0; s < 4; ++s) {
        const int coln = tile_n0 + s * 16 + l16;
        float bb = bias ? bias[coln] : 0.f;
        #pragma unroll
        for (int r = 0; r < 8; ++r) {
            int row = tile_m + r + (half ? 8 : 0);
            float vv = acc[s][r] + bb;
            if (act == 1) vv = 0.5f * vv * (1.f + erff(vv * 0.70710678118654752f));
            size_t o = (size_t)row * ldc + coln;
            if (Cf) Cf[o] = vv;
            if (Cb) Cb[o] = f2bf(vv);
        }
    }
}

// ---------------------------------------------------------------------------
// FAVOR+ feature (in-place f32 + bf16 copy):
//   e = exp(G[m] - 0.0625*sum_d q_d^2); G[m]=e; Gb[m]=bf16(e)
// ---------------------------------------------------------------------------
__global__ void favor_feature(float* __restrict__ G, const float* __restrict__ qraw,
                              bf16_t* __restrict__ Gb)
{
    int idx = blockIdx.x * blockDim.x + threadIdx.x;
    if (idx >= NTOK * HH) return;
    int bt = idx >> 3, h = idx & 7;
    const float* qp = qraw + (size_t)bt * DD + h * HDm;
    float ss = 0.f;
    #pragma unroll 8
    for (int d = 0; d < HDm; ++d) { float u = qp[d]; ss += u * u; }
    float c = 0.0625f * ss;
    size_t o = (size_t)bt * DD + h * HDm;
    float* g = G + o;
    bf16_t* gb = Gb + o;
    #pragma unroll 8
    for (int m = 0; m < HDm; ++m) {
        float e = expf(g[m] - c);
        g[m]  = e;
        gb[m] = f2bf(e);
    }
}

// ---------------------------------------------------------------------------
// Pass A: per (b,h,chunk): KVt[d][m] = sum_s v[s][d]*kf[s][m] (WMMA)
// and ksum[m] = sum_s kf[s][m].  grid = BB*HH*NC, block = 128.
// ---------------------------------------------------------------------------
__global__ __launch_bounds__(128)
void chunk_partial(const float* __restrict__ kf, const float* __restrict__ v,
                   float* __restrict__ kvt, float* __restrict__ ksum)
{
    int blk = blockIdx.x;
    int bh  = blk >> 5, c = blk & (NC - 1);
    int b   = bh >> 3,  h = bh & 7;
    __shared__ __attribute__((aligned(16))) bf16_t Vt[CH * CH];  // [d][s]
    __shared__ __attribute__((aligned(16))) bf16_t Kt[CH * CH];  // [m][s]
    int tid = threadIdx.x;
    size_t base = ((size_t)b * TT + (size_t)c * CH) * DD + (size_t)h * HDm;

    for (int i = tid; i < CH * CH; i += 128) {
        int s = i >> 6, col = i & 63;          // coalesced over col
        Vt[col * CH + s] = f2bf(v[base + (size_t)s * DD + col]);
        Kt[col * CH + s] = f2bf(kf[base + (size_t)s * DD + col]);
    }
    __syncthreads();

    int lane = tid & 31, wave = tid >> 5, half = lane >> 4, l16 = lane & 15;
    int tm = wave * 16;                         // d-strip
    v8f acc[4];
    #pragma unroll
    for (int s4 = 0; s4 < 4; ++s4) acc[s4] = (v8f){0.f,0.f,0.f,0.f,0.f,0.f,0.f,0.f};
    #pragma unroll
    for (int kb = 0; kb < 2; ++kb) {
        v16bf av = a_op(&Vt[(tm + l16) * CH], kb << 5, half);
        #pragma unroll
        for (int s4 = 0; s4 < 4; ++s4) {
            v16bf bv = b_op(&Kt[(s4 * 16 + l16) * CH], kb << 5, half);
            acc[s4] = WMMA_BF16(av, bv, acc[s4]);
        }
    }
    float* outp = kvt + (size_t)blk * 4096;
    #pragma unroll
    for (int s4 = 0; s4 < 4; ++s4) {
        int mcol = s4 * 16 + l16;
        #pragma unroll
        for (int r = 0; r < 8; ++r) {
            int drow = tm + r + (half ? 8 : 0);
            outp[drow * 64 + mcol] = acc[s4][r];
        }
    }
    if (tid < 64) {
        float ssum = 0.f;
        #pragma unroll 8
        for (int s = 0; s < CH; ++s) ssum += (float)Kt[tid * CH + s];
        ksum[(size_t)blk * 64 + tid] = ssum;
    }
}

// ---------------------------------------------------------------------------
// Pass B: chunk-level exclusive prefix over NC chunks. grid = BB*HH, block 256.
// ---------------------------------------------------------------------------
__global__ __launch_bounds__(256)
void chunk_scan(const float* __restrict__ kvt, const float* __restrict__ ksum,
                bf16_t* __restrict__ sprev, float* __restrict__ zprev)
{
    int bh = blockIdx.x, tid = threadIdx.x;
    float S[16];
    #pragma unroll
    for (int i = 0; i < 16; ++i) S[i] = 0.f;
    float z = 0.f;
    for (int c = 0; c < NC; ++c) {
        size_t blk = (size_t)(bh * NC + c);
        #pragma unroll
        for (int i = 0; i < 16; ++i) {
            size_t o = blk * 4096 + (size_t)tid * 16 + i;
            sprev[o] = f2bf(S[i]);
            S[i] += kvt[o];
        }
        if (tid < 64) {
            zprev[blk * 64 + tid] = z;
            z += ksum[blk * 64 + tid];
        }
    }
}

// ---------------------------------------------------------------------------
// Pass C: within-chunk causal attention; Qr/Kr staged with async global->LDS.
//   P = tril(Qf @ Kf^T); den_t = rowsum(P) + Qf.z_prev
//   num = P @ V + Qf @ S_prev ; out = num / (den + 1e-16)
// grid = BB*HH*NC, block = 128 (4 waves, wave = 16-token strip).
// ---------------------------------------------------------------------------
__global__ __launch_bounds__(128)
void chunk_attn(const bf16_t* __restrict__ qfb, const bf16_t* __restrict__ kfb,
                const float* __restrict__ v, const bf16_t* __restrict__ sprev,
                const float* __restrict__ zprev, float* __restrict__ out)
{
    int blk = blockIdx.x;
    int bh  = blk >> 5, c = blk & (NC - 1);
    int b   = bh >> 3,  h = bh & 7;
    __shared__ __attribute__((aligned(16))) bf16_t Qr[CH * CH];  // [t][m]
    __shared__ __attribute__((aligned(16))) bf16_t Kr[CH * CH];  // [s][m]
    __shared__ __attribute__((aligned(16))) bf16_t Vt[CH * CH];  // [d][s]
    __shared__ __attribute__((aligned(16))) bf16_t Pr[CH * CH];  // [t][s]
    __shared__ float den[CH];
    __shared__ float zp[CH];
    int tid = threadIdx.x;
    size_t base = ((size_t)b * TT + (size_t)c * CH) * DD + (size_t)h * HDm;

    // Qr/Kr are consumed in the same row-major [token][m] layout they have in
    // memory -> async DMA straight into LDS (16B per lane, ASYNCcnt-tracked).
    unsigned qr_lds = (unsigned)(size_t)&Qr[0];
    unsigned kr_lds = (unsigned)(size_t)&Kr[0];
    #pragma unroll
    for (int it = 0; it < 4; ++it) {
        int g  = tid + it * 128;               // 512 x 16B chunks per tile
        int rr = g >> 3;
        int cl = (g & 7) * 8;
        size_t go = base + (size_t)rr * DD + cl;
        async_cp_b128(qr_lds + g * 16, (const void*)(qfb + go));
        async_cp_b128(kr_lds + g * 16, (const void*)(kfb + go));
    }
    // V needs a transpose -> scalar staging (overlaps the async copies)
    for (int i = tid; i < CH * CH; i += 128) {
        int rr = i >> 6, cl = i & 63;
        Vt[cl * CH + rr] = f2bf(v[base + (size_t)rr * DD + cl]);
    }
    if (tid < 64) zp[tid] = zprev[(size_t)blk * 64 + tid];
    async_wait0();
    __syncthreads();

    int lane = tid & 31, wave = tid >> 5, half = lane >> 4, l16 = lane & 15;
    int tm = wave * 16;                         // token strip

    // --- P = Qr @ Kr^T ---
    v8f pacc[4];
    #pragma unroll
    for (int s4 = 0; s4 < 4; ++s4) pacc[s4] = (v8f){0.f,0.f,0.f,0.f,0.f,0.f,0.f,0.f};
    #pragma unroll
    for (int kb = 0; kb < 2; ++kb) {
        v16bf av = a_op(&Qr[(tm + l16) * CH], kb << 5, half);
        #pragma unroll
        for (int s4 = 0; s4 < 4; ++s4) {
            v16bf bv = b_op(&Kr[(s4 * 16 + l16) * CH], kb << 5, half);
            pacc[s4] = WMMA_BF16(av, bv, pacc[s4]);
        }
    }
    #pragma unroll
    for (int s4 = 0; s4 < 4; ++s4) {
        int scol = s4 * 16 + l16;
        #pragma unroll
        for (int r = 0; r < 8; ++r) {
            int trow = tm + r + (half ? 8 : 0);
            float pv = (scol <= trow) ? pacc[s4][r] : 0.f;
            Pr[trow * CH + scol] = f2bf(pv);
        }
    }
    __syncthreads();

    // --- den ---
    if (tid < 64) {
        float dd_ = 0.f;
        #pragma unroll 8
        for (int s = 0; s < CH; ++s) dd_ += (float)Pr[tid * CH + s];
        #pragma unroll 8
        for (int m = 0; m < CH; ++m) dd_ += (float)Qr[tid * CH + m] * zp[m];
        den[tid] = dd_;
    }
    __syncthreads();

    // --- num = Pr @ V + Qr @ S_prev ---
    const bf16_t* sp = sprev + (size_t)blk * 4096;  // [d][m]
    v8f acc[4];
    #pragma unroll
    for (int d4 = 0; d4 < 4; ++d4) acc[d4] = (v8f){0.f,0.f,0.f,0.f,0.f,0.f,0.f,0.f};
    #pragma unroll
    for (int kb = 0; kb < 2; ++kb) {
        v16bf ap = a_op(&Pr[(tm + l16) * CH], kb << 5, half);
        #pragma unroll
        for (int d4 = 0; d4 < 4; ++d4) {
            v16bf bv = b_op(&Vt[(d4 * 16 + l16) * CH], kb << 5, half);
            acc[d4] = WMMA_BF16(ap, bv, acc[d4]);
        }
    }
    #pragma unroll
    for (int kb = 0; kb < 2; ++kb) {
        v16bf aq = a_op(&Qr[(tm + l16) * CH], kb << 5, half);
        #pragma unroll
        for (int d4 = 0; d4 < 4; ++d4) {
            v16bf bv = b_op(sp + (size_t)(d4 * 16 + l16) * 64, kb << 5, half);
            acc[d4] = WMMA_BF16(aq, bv, acc[d4]);
        }
    }
    #pragma unroll
    for (int d4 = 0; d4 < 4; ++d4) {
        int dcol = d4 * 16 + l16;
        #pragma unroll
        for (int r = 0; r < 8; ++r) {
            int trow = tm + r + (half ? 8 : 0);
            out[base + (size_t)trow * DD + dcol] = acc[d4][r] / (den[trow] + 1e-16f);
        }
    }
}

// ---------------------------------------------------------------------------
// x = x + LayerNorm(a)*g + b ; refresh bf16 copy of x
// ---------------------------------------------------------------------------
__global__ __launch_bounds__(128)
void ln_res(const float* __restrict__ a, float* __restrict__ x, bf16_t* __restrict__ xb,
            const float* __restrict__ g, const float* __restrict__ bvec)
{
    int row = blockIdx.x, tid = threadIdx.x;
    float4 vv = ((const float4*)(a + (size_t)row * DD))[tid];
    float s  = vv.x + vv.y + vv.z + vv.w;
    float ss = vv.x * vv.x + vv.y * vv.y + vv.z * vv.z + vv.w * vv.w;
    __shared__ float r1[128], r2[128];
    r1[tid] = s; r2[tid] = ss;
    __syncthreads();
    for (int st = 64; st > 0; st >>= 1) {
        if (tid < st) { r1[tid] += r1[tid + st]; r2[tid] += r2[tid + st]; }
        __syncthreads();
    }
    float mu   = r1[0] * (1.f / DD);
    float var  = r2[0] * (1.f / DD) - mu * mu;
    float rinv = rsqrtf(var + 1e-5f);
    float vals[4] = {vv.x, vv.y, vv.z, vv.w};
    #pragma unroll
    for (int j = 0; j < 4; ++j) {
        int cc = tid * 4 + j;
        float y = (vals[j] - mu) * rinv * g[cc] + bvec[cc];
        size_t o = (size_t)row * DD + cc;
        float xn = x[o] + y;
        x[o]  = xn;
        xb[o] = f2bf(xn);
    }
}

// ---------------------------------------------------------------------------
// Host-side orchestration
// ---------------------------------------------------------------------------
extern "C" void kernel_launch(void* const* d_in, const int* in_sizes, int n_in,
                              void* d_out, int out_size, void* d_ws, size_t ws_size,
                              hipStream_t stream)
{
    (void)in_sizes; (void)n_in; (void)out_size; (void)ws_size;

    const int*   tokens = (const int*)  d_in[0];
    const float* emb    = (const float*)d_in[1];
    const float* Wq     = (const float*)d_in[2];
    const float* bq     = (const float*)d_in[3];
    const float* Wk     = (const float*)d_in[4];
    const float* bk     = (const float*)d_in[5];
    const float* Wv     = (const float*)d_in[6];
    const float* bv     = (const float*)d_in[7];
    const float* rfs    = (const float*)d_in[8];
    const float* ln1_g  = (const float*)d_in[9];
    const float* ln1_b  = (const float*)d_in[10];
    const float* ln2_g  = (const float*)d_in[11];
    const float* ln2_b  = (const float*)d_in[12];
    const float* WU     = (const float*)d_in[13];
    const float* bU     = (const float*)d_in[14];
    const float* WVm    = (const float*)d_in[15];
    const float* bV     = (const float*)d_in[16];
    const float* Wout   = (const float*)d_in[17];
    const float* bout   = (const float*)d_in[18];

    char* ws = (char*)d_ws;
    const size_t MB = (size_t)1 << 20;
    size_t off = 0;
    auto alloc = [&](size_t bytes) -> char* {
        char* p = ws + off;
        off += (bytes + 255) & ~(size_t)255;
        return p;
    };

    bf16_t* wqB  = (bf16_t*)alloc((size_t)LL * DD * DD * 2);
    bf16_t* wkB  = (bf16_t*)alloc((size_t)LL * DD * DD * 2);
    bf16_t* wvB  = (bf16_t*)alloc((size_t)LL * DD * DD * 2);
    bf16_t* wuB  = (bf16_t*)alloc((size_t)LL * DD * FF * 2);
    bf16_t* wv2B = (bf16_t*)alloc((size_t)LL * FF * DD * 2);
    bf16_t* woB  = (bf16_t*)alloc((size_t)DD * VV * 2);
    bf16_t* rfB  = (bf16_t*)alloc((size_t)LL * HH * HDm * HDm * 2);
    float*  x    = (float*) alloc((size_t)NTOK * DD * 4);
    bf16_t* xb   = (bf16_t*)alloc((size_t)NTOK * DD * 2);
    char* scr = alloc(64 * MB);
    // attention phase
    float*  q     = (float*) (scr +  0 * MB);
    bf16_t* qb    = (bf16_t*)(scr +  8 * MB);
    float*  kbuf  = (float*) (scr + 12 * MB);
    bf16_t* kb    = (bf16_t*)(scr + 20 * MB);
    float*  vbuf  = (float*) (scr + 24 * MB);
    float*  Gq    = (float*) (scr + 32 * MB);
    float*  Gk    = (float*) (scr + 40 * MB);
    float*  attn  = (float*) (scr + 48 * MB);
    bf16_t* Gqb   = (bf16_t*)(scr + 56 * MB);          // bf16 feature copies
    bf16_t* Gkb   = (bf16_t*)(scr + 60 * MB);
    // chunked-scan phase (aliases q/qb/kbuf which are dead by then)
    float*  kvt   = (float*) (scr +  0 * MB);          // 512*4096 f32 = 8MB
    bf16_t* spv   = (bf16_t*)(scr +  8 * MB);          // 512*4096 bf16 = 4MB
    float*  ksumB = (float*) (scr + 12 * MB);          // 128KB
    float*  zprvB = (float*) (scr + 12 * MB + 262144); // 128KB
    // FFN phase (aliases everything below attn)
    bf16_t* hb    = (bf16_t*)(scr +  0 * MB);
    float*  f2    = (float*) (scr + 16 * MB);

    auto pk = [&](const float* sP, bf16_t* dP, int K, int N, float sc) {
        int n = K * N;
        pack_w<<<dim3((n + 255) / 256), dim3(256), 0, stream>>>(sP, dP, K, N, sc);
    };
    for (int l = 0; l < LL; ++l) {
        pk(Wq  + (size_t)l * DD * DD, wqB  + (size_t)l * DD * DD, DD, DD, 1.f);
        pk(Wk  + (size_t)l * DD * DD, wkB  + (size_t)l * DD * DD, DD, DD, 1.f);
        pk(Wv  + (size_t)l * DD * DD, wvB  + (size_t)l * DD * DD, DD, DD, 1.f);
        pk(WU  + (size_t)l * DD * FF, wuB  + (size_t)l * DD * FF, DD, FF, 1.f);
        pk(WVm + (size_t)l * FF * DD, wv2B + (size_t)l * FF * DD, FF, DD, 1.f);
        for (int h = 0; h < HH; ++h) {
            size_t o = (size_t)(l * HH + h) * HDm * HDm;
            pk(rfs + o, rfB + o, HDm, HDm, 0.35355339059f); // HD^{-1/4} folded in
        }
    }
    pk(Wout, woB, DD, VV, 1.f);

    embed_pos<<<dim3(NTOK * DD / 256), dim3(256), 0, stream>>>(tokens, emb, x, xb);

    auto gemm = [&](const bf16_t* A, int lda, const bf16_t* Bp,
                    const float* bias, float* Cf, bf16_t* Cb, int ldc,
                    int M, int N, int K, int act) {
        gemm_bf16_wmma<<<dim3(N / 64, M / 64), dim3(128), 0, stream>>>(
            A, lda, Bp, bias, Cf, Cb, ldc, K, act);
    };

    for (int l = 0; l < LL; ++l) {
        const bf16_t* wq_l = wqB  + (size_t)l * DD * DD;
        const bf16_t* wk_l = wkB  + (size_t)l * DD * DD;
        const bf16_t* wv_l = wvB  + (size_t)l * DD * DD;
        const bf16_t* wu_l = wuB  + (size_t)l * DD * FF;
        const bf16_t* w2_l = wv2B + (size_t)l * FF * DD;

        gemm(xb, DD, wq_l, bq + l * DD, q,    qb,      DD, NTOK, DD, DD, 0);
        gemm(xb, DD, wk_l, bk + l * DD, kbuf, kb,      DD, NTOK, DD, DD, 0);
        gemm(xb, DD, wv_l, bv + l * DD, vbuf, nullptr, DD, NTOK, DD, DD, 0);

        for (int h = 0; h < HH; ++h) {
            const bf16_t* rf = rfB + (size_t)(l * HH + h) * HDm * HDm;
            gemm(qb + h * HDm, DD, rf, nullptr, Gq + h * HDm, nullptr, DD, NTOK, HDm, HDm, 0);
            gemm(kb + h * HDm, DD, rf, nullptr, Gk + h * HDm, nullptr, DD, NTOK, HDm, HDm, 0);
        }
        favor_feature<<<dim3(NTOK * HH / 256), dim3(256), 0, stream>>>(Gq, q,    Gqb);
        favor_feature<<<dim3(NTOK * HH / 256), dim3(256), 0, stream>>>(Gk, kbuf, Gkb);

        // chunked causal linear attention (WMMA passes + 32-step chunk scan)
        chunk_partial<<<dim3(BB * HH * NC), dim3(128), 0, stream>>>(Gk, vbuf, kvt, ksumB);
        chunk_scan<<<dim3(BB * HH), dim3(256), 0, stream>>>(kvt, ksumB, spv, zprvB);
        chunk_attn<<<dim3(BB * HH * NC), dim3(128), 0, stream>>>(Gqb, Gkb, vbuf, spv, zprvB, attn);

        ln_res<<<dim3(NTOK), dim3(128), 0, stream>>>(attn, x, xb, ln1_g + l * DD, ln1_b + l * DD);

        gemm(xb, DD, wu_l, bU + l * FF, nullptr, hb, FF, NTOK, FF, DD, 1);
        gemm(hb, FF, w2_l, bV + l * DD, f2, nullptr, DD, NTOK, DD, FF, 0);
        ln_res<<<dim3(NTOK), dim3(128), 0, stream>>>(f2, x, xb, ln2_g + l * DD, ln2_b + l * DD);
    }

    gemm(xb, DD, woB, bout, (float*)d_out, nullptr, VV, NTOK, VV, DD, 0);
}
